// TextEncoder_20804821582371
// MI455X (gfx1250) — matrix operand order repaired and checked by reading, not compiled, not run on hardware
//
#include <hip/hip_runtime.h>
#include <cstdint>
#include <cstddef>

#define B_    4
#define S_    2048
#define N_    2049      // S+1
#define D_    512
#define H_    8
#define DH_   64
#define L_    6
#define FF_   2048
#define FUSED_ 4736
#define MTOK  8196      // B*(S+1)
#define MP    8320      // MTOK padded to 65 tiles of 128
#define QP    2064      // query rows padded (129*16)
#define KP    2080      // key rows padded (65*32)
#define QT    129
#define NEG_  (-3.0e38f)

typedef __bf16 bf16;
typedef __attribute__((ext_vector_type(16))) __bf16 v16bf;
typedef __attribute__((ext_vector_type(8)))  __bf16 v8bf;
typedef __attribute__((ext_vector_type(8)))  float  v8f;

__device__ __forceinline__ v16bf pack16(v8bf lo, v8bf hi) {
  v16bf r;
#pragma unroll
  for (int e = 0; e < 8; ++e) { r[e] = lo[e]; r[e + 8] = hi[e]; }
  return r;
}
__device__ __forceinline__ v8bf ld8(const bf16* p) { return *(const v8bf*)p; }

// ---------------- elementwise / setup kernels ----------------

__global__ void k_f32_to_bf16(const float* __restrict__ src, bf16* __restrict__ dst, int count) {
  int i = blockIdx.x * blockDim.x + threadIdx.x;
  if (i < count) dst[i] = (bf16)src[i];
}

__global__ void k_rope_tables(float* __restrict__ cosT, float* __restrict__ sinT) {
  int i = blockIdx.x * blockDim.x + threadIdx.x;
  if (i >= N_ * 32) return;
  int p = i >> 5, d = i & 31;
  float inv = __expf(-(float)d * (logf(10000.0f) / 32.0f));  // 10000^(-d/32)
  float ang = (float)p * inv;
  cosT[i] = cosf(ang);
  sinT[i] = sinf(ang);
}

__global__ void k_embed(const int* __restrict__ text, const float* __restrict__ emb,
                        const float* __restrict__ cls, float* __restrict__ x) {
  int i = blockIdx.x * blockDim.x + threadIdx.x;
  if (i >= MTOK * D_) return;
  int row = i / D_, d = i % D_;
  int b = row / N_, t = row % N_;
  x[i] = (t < S_) ? emb[(size_t)text[b * S_ + t] * D_ + d] : cls[d];
}

// LayerNorm (one wave per row) -> bf16
__global__ __launch_bounds__(256) void k_ln_bf16(const float* __restrict__ x,
    const float* __restrict__ g, const float* __restrict__ bt, bf16* __restrict__ out) {
  int wid = threadIdx.x >> 5, lane = threadIdx.x & 31;
  int row = blockIdx.x * 8 + wid;
  if (row >= MTOK) return;
  const float* xr = x + (size_t)row * D_;
  float v[16], s = 0.f, s2 = 0.f;
#pragma unroll
  for (int i = 0; i < 16; ++i) { float t = xr[lane * 16 + i]; v[i] = t; s += t; s2 += t * t; }
#pragma unroll
  for (int m = 1; m < 32; m <<= 1) { s += __shfl_xor(s, m); s2 += __shfl_xor(s2, m); }
  float mean = s * (1.0f / D_);
  float var  = s2 * (1.0f / D_) - mean * mean;
  float inv  = rsqrtf(var + 1e-5f);
  bf16* orow = out + (size_t)row * D_;
#pragma unroll
  for (int i = 0; i < 16; ++i) {
    int c = lane * 16 + i;
    orow[c] = (bf16)((v[i] - mean) * inv * g[c] + bt[c]);
  }
}

// ---------------- WMMA GEMM: C = A(MxK bf16) * B(KxN bf16) ----------------
// MODE 0: store fp32, 1: store bf16, 2: fp32 accumulate (C += ...)
// Block: 256 thr = 8 waves (2M x 4N), block tile 128x128, BK=32, wave tile 64x32.
template <int MODE>
__global__ __launch_bounds__(256) void k_gemm(const bf16* __restrict__ A,
    const bf16* __restrict__ Bm, void* __restrict__ Cout,
    int M, int N, int K) {
  __shared__ __align__(16) bf16 sA[128 * 40];
  __shared__ __align__(16) bf16 sBT[128 * 40];  // B stored transposed [n][k]
  int tid = threadIdx.x;
  int wid = tid >> 5, lane = tid & 31;
  int wm = wid >> 2, wn = wid & 3;
  int m15 = lane & 15, kh = lane >> 4;
  int bN = blockIdx.x, bM = blockIdx.y;
  v8f acc[4][2];
#pragma unroll
  for (int i = 0; i < 4; ++i)
#pragma unroll
    for (int j = 0; j < 2; ++j)
#pragma unroll
      for (int e = 0; e < 8; ++e) acc[i][j][e] = 0.f;

  const size_t aRow0 = (size_t)bM * 128;
  const size_t bCol0 = (size_t)bN * 128;
  for (int k0 = 0; k0 < K; k0 += 32) {
    // prefetch next K tile (global_prefetch_b8); weights mostly hit L2
    if (k0 + 32 < K) {
      __builtin_prefetch(A + (aRow0 + (tid >> 1)) * K + k0 + 32 + (tid & 1) * 16, 0, 3);
      __builtin_prefetch(Bm + (size_t)(k0 + 32 + (tid >> 4)) * N + bCol0 + (tid & 15) * 8, 0, 3);
    }
    // A tile 128x32 (row-major in LDS): 2 x 16B chunks per thread
#pragma unroll
    for (int it = 0; it < 2; ++it) {
      int c = tid * 2 + it;
      int r = c >> 2, ch = (c & 3) * 8;
      *(uint4*)&sA[r * 40 + ch] =
          *(const uint4*)(A + (aRow0 + r) * K + k0 + ch);
    }
    // B tile 32x128, transposed into LDS
#pragma unroll
    for (int it = 0; it < 2; ++it) {
      int c = tid * 2 + it;
      int r = c >> 4, co = (c & 15) * 8;
      union { uint4 u; bf16 h[8]; } tmp;
      tmp.u = *(const uint4*)(Bm + (size_t)(k0 + r) * N + bCol0 + co);
#pragma unroll
      for (int e = 0; e < 8; ++e) sBT[(co + e) * 40 + r] = tmp.h[e];
    }
    __syncthreads();
    v16bf af[4], bfr[2];
#pragma unroll
    for (int i = 0; i < 4; ++i) {  // A frag: lane-half selects K 0-7/16-23 vs 8-15/24-31
      const bf16* p = &sA[(wm * 64 + i * 16 + m15) * 40 + 8 * kh];
      af[i] = pack16(ld8(p), ld8(p + 16));
    }
#pragma unroll
    for (int j = 0; j < 2; ++j) {  // B frag: lane-half selects K 0-7/16-23 vs 8-15/24-31
      const bf16* p = &sBT[(wn * 32 + j * 16 + m15) * 40 + 8 * kh];
      bfr[j] = pack16(ld8(p), ld8(p + 16));
    }
#pragma unroll
    for (int i = 0; i < 4; ++i)
#pragma unroll
      for (int j = 0; j < 2; ++j)
        acc[i][j] = __builtin_amdgcn_wmma_f32_16x16x32_bf16(
            false, af[i], false, bfr[j], (short)0, acc[i][j], false, false);
    __syncthreads();
  }

  const int rowBase = bM * 128 + wm * 64 + 8 * kh;
  const int colBase = bN * 128 + wn * 32 + m15;
  const bool fullM = (bM * 128 + 128 <= M);   // block-uniform fast path
  if (fullM) {
#pragma unroll
    for (int i = 0; i < 4; ++i)
#pragma unroll
      for (int j = 0; j < 2; ++j) {
        size_t base = (size_t)(rowBase + i * 16) * N + colBase + j * 16;
#pragma unroll
        for (int r = 0; r < 8; ++r) {
          size_t off = base + (size_t)r * N;
          if (MODE == 0)      ((float*)Cout)[off] = acc[i][j][r];
          else if (MODE == 1) ((bf16*)Cout)[off]  = (bf16)acc[i][j][r];
          else                ((float*)Cout)[off] += acc[i][j][r];
        }
      }
  } else {
#pragma unroll
    for (int i = 0; i < 4; ++i)
#pragma unroll
      for (int j = 0; j < 2; ++j) {
#pragma unroll
        for (int r = 0; r < 8; ++r) {
          int row = rowBase + i * 16 + r;
          if (row < M) {
            size_t off = (size_t)row * N + colBase + j * 16;
            if (MODE == 0)      ((float*)Cout)[off] = acc[i][j][r];
            else if (MODE == 1) ((bf16*)Cout)[off]  = (bf16)acc[i][j][r];
            else                ((float*)Cout)[off] += acc[i][j][r];
          }
        }
      }
  }
}

// ---------------- split proj -> q(rope*scale), k(rope), vT, silu-gated ff ----------------
__global__ __launch_bounds__(256) void k_post_proj(const bf16* __restrict__ proj,
    bf16* __restrict__ q, bf16* __restrict__ k, bf16* __restrict__ vT,
    bf16* __restrict__ ffh, const float* __restrict__ cosT, const float* __restrict__ sinT) {
  int row = blockIdx.x;                 // 0..MTOK-1
  int b = row / N_, t = row % N_;
  const bf16* pr = proj + (size_t)row * FUSED_;
  const float* cs = cosT + t * 32;
  const float* sn = sinT + t * 32;
  // rope for 8 q heads + shared k: 9 groups * 32 pairs
  for (int task = threadIdx.x; task < 288; task += 256) {
    int g = task >> 5, d = task & 31;
    int base = (g < 8) ? g * 64 : 512;
    float a = (float)pr[base + d], bb = (float)pr[base + d + 32];
    float c = cs[d], s = sn[d];
    float o1 = a * c - bb * s;
    float o2 = bb * c + a * s;
    if (g < 8) {
      size_t qb = ((size_t)(b * H_ + g) * QP + t) * 64;
      q[qb + d]      = (bf16)(o1 * 0.125f);   // DH^-0.5
      q[qb + d + 32] = (bf16)(o2 * 0.125f);
    } else {
      size_t kb = ((size_t)b * KP + t) * 64;
      k[kb + d]      = (bf16)o1;
      k[kb + d + 32] = (bf16)o2;
    }
  }
  for (int d = threadIdx.x; d < 64; d += 256)
    vT[((size_t)b * 64 + d) * KP + t] = pr[576 + d];
  for (int c = threadIdx.x; c < FF_; c += 256) {
    float xg = (float)pr[2688 + c];           // ff_gate
    float xv = (float)pr[640 + c];            // ff_x
    float sg = xg / (1.0f + __expf(-xg));     // silu
    ffh[(size_t)row * FF_ + c] = (bf16)(sg * xv);
  }
}

// ---------------- flash attention, one wave per (b,h,16-row query tile) ----------------
__global__ __launch_bounds__(256) void k_attn(const bf16* __restrict__ q,
    const bf16* __restrict__ k, const bf16* __restrict__ vT,
    const int* __restrict__ text, bf16* __restrict__ o) {
  __shared__ __align__(16) bf16 sP[8][16 * 40];  // per-wave P relayout scratch
  int wid = threadIdx.x >> 5, lane = threadIdx.x & 31;
  int gid = blockIdx.x * 8 + wid;
  if (gid >= B_ * H_ * QT) return;               // wave-uniform
  int qt = gid % QT;
  int h  = (gid / QT) % H_;
  int b  = gid / (QT * H_);
  int q0 = qt * 16;
  int m15 = lane & 15, kh = lane >> 4;

  v16bf qf[2];
  const bf16* qp = q + ((size_t)(b * H_ + h) * QP + q0 + m15) * 64;
#pragma unroll
  for (int kk = 0; kk < 2; ++kk) {               // A frag over dh = kk*32..+31
    const bf16* p = qp + kk * 32 + 8 * kh;
    qf[kk] = pack16(ld8(p), ld8(p + 16));
  }
  v16bf onesB;                                    // all-ones B frag: rowsum(P) = P x 1
#pragma unroll
  for (int e = 0; e < 16; ++e) onesB[e] = (bf16)1.0f;

  float m_s[8], l_s[8];
  v8f oacc[4];
#pragma unroll
  for (int r = 0; r < 8; ++r) { m_s[r] = NEG_; l_s[r] = 0.f; }
#pragma unroll
  for (int d4 = 0; d4 < 4; ++d4)
#pragma unroll
    for (int r = 0; r < 8; ++r) oacc[d4][r] = 0.f;

  int qmax = (q0 + 15 < S_) ? (q0 + 15) : S_;    // max valid query index (n-1 == S_)
  int nblocks = qmax / 32 + 1;
  bool needCls = (q0 + 15 >= S_);                // tile containing the cls row
  for (int jb = 0; jb < nblocks; ++jb) {
    int j0 = jb * 32;
    v8f s[2];
    float p0[2][8];
#pragma unroll
    for (int nb = 0; nb < 2; ++nb) {
#pragma unroll
      for (int e = 0; e < 8; ++e) s[nb][e] = 0.f;
      int key = j0 + nb * 16 + m15;
#pragma unroll
      for (int kk = 0; kk < 2; ++kk) {           // B frag = K^T, table-order dh per lane
        const bf16* p = k + ((size_t)b * KP + key) * 64 + kk * 32 + 8 * kh;
        v16bf kf = pack16(ld8(p), ld8(p + 16));
        s[nb] = __builtin_amdgcn_wmma_f32_16x16x32_bf16(
            false, qf[kk], false, kf, (short)0, s[nb], false, false);
      }
      bool clsAllowed = true;
      if (needCls) clsAllowed = (key < S_) ? (text[b * S_ + key] != 0) : (key == S_);
#pragma unroll
      for (int r = 0; r < 8; ++r) {
        int qrow = q0 + r + 8 * kh;
        bool ok = (key <= qrow) && (key < N_) && (qrow != S_ || clsAllowed);
        if (!ok) s[nb][r] = NEG_;
      }
    }
    // online softmax: row max reduced across the 16-lane groups (shuffle tree)
    float mb[8];
#pragma unroll
    for (int r = 0; r < 8; ++r) mb[r] = fmaxf(s[0][r], s[1][r]);
#pragma unroll
    for (int msk = 1; msk < 16; msk <<= 1)
#pragma unroll
      for (int r = 0; r < 8; ++r) mb[r] = fmaxf(mb[r], __shfl_xor(mb[r], msk));
    float alpha[8];
#pragma unroll
    for (int r = 0; r < 8; ++r) {
      float mn = fmaxf(m_s[r], mb[r]);
      alpha[r] = __expf(m_s[r] - mn);
      m_s[r] = mn;
    }
#pragma unroll
    for (int r = 0; r < 8; ++r) {
      p0[0][r] = __expf(s[0][r] - m_s[r]);
      p0[1][r] = __expf(s[1][r] - m_s[r]);
      l_s[r] *= alpha[r];
    }
#pragma unroll
    for (int d4 = 0; d4 < 4; ++d4)
#pragma unroll
      for (int r = 0; r < 8; ++r) oacc[d4][r] *= alpha[r];
    // P: C-layout -> A-layout through per-wave LDS (DS pipe is in-order per wave)
#pragma unroll
    for (int nb = 0; nb < 2; ++nb)
#pragma unroll
      for (int r = 0; r < 8; ++r)
        sP[wid][(r + 8 * kh) * 40 + nb * 16 + m15] = (bf16)p0[nb][r];
    __asm__ volatile("" ::: "memory");
    v16bf pf;
    {
      const bf16* pp = &sP[wid][m15 * 40 + 8 * kh];
      pf = pack16(ld8(pp), ld8(pp + 16));
    }
    __asm__ volatile("" ::: "memory");
    // row-sum of P via WMMA against ones (replaces a 4-step shuffle tree);
    // C layout rows r+8*kh match l_s[] indexing exactly.
    {
      v8f lsv;
#pragma unroll
      for (int e = 0; e < 8; ++e) lsv[e] = 0.f;
      lsv = __builtin_amdgcn_wmma_f32_16x16x32_bf16(
          false, pf, false, onesB, (short)0, lsv, false, false);
#pragma unroll
      for (int r = 0; r < 8; ++r) l_s[r] += lsv[r];
    }
#pragma unroll
    for (int d4 = 0; d4 < 4; ++d4) {             // O += P * V, B frag from vT (table-order keys)
      int dh = d4 * 16 + m15;
      const bf16* vp = vT + ((size_t)b * 64 + dh) * KP + j0 + 8 * kh;
      v16bf vf = pack16(ld8(vp), ld8(vp + 16));
      oacc[d4] = __builtin_amdgcn_wmma_f32_16x16x32_bf16(
          false, pf, false, vf, (short)0, oacc[d4], false, false);
    }
  }
#pragma unroll
  for (int d4 = 0; d4 < 4; ++d4)
#pragma unroll
    for (int r = 0; r < 8; ++r) {
      int qrow = q0 + r + 8 * kh;
      if (qrow < N_) {
        float val = oacc[d4][r] / l_s[r];
        o[(size_t)(b * N_ + qrow) * (H_ * DH_) + h * 64 + d4 * 16 + m15] = (bf16)val;
      }
    }
}

// ---------------- finalize ----------------
__global__ void k_copy_tokens(const float* __restrict__ x, float* __restrict__ out) {
  int i = blockIdx.x * blockDim.x + threadIdx.x;
  if (i >= B_ * S_ * D_) return;
  int d = i % D_;
  int t = (i / D_) % S_;
  int b = i / (S_ * D_);
  out[i] = x[((size_t)b * N_ + t) * D_ + d];
}

__global__ void k_cls_ln(const float* __restrict__ x, const float* __restrict__ g,
                         const float* __restrict__ bt, float* __restrict__ out) {
  int wid = threadIdx.x >> 5, lane = threadIdx.x & 31;
  int b = wid;
  if (b >= B_) return;
  const float* xr = x + ((size_t)b * N_ + S_) * D_;
  float v[16], s = 0.f, s2 = 0.f;
#pragma unroll
  for (int i = 0; i < 16; ++i) { float t = xr[lane * 16 + i]; v[i] = t; s += t; s2 += t * t; }
#pragma unroll
  for (int m = 1; m < 32; m <<= 1) { s += __shfl_xor(s, m); s2 += __shfl_xor(s2, m); }
  float mean = s * (1.0f / D_);
  float var  = s2 * (1.0f / D_) - mean * mean;
  float inv  = rsqrtf(var + 1e-5f);
#pragma unroll
  for (int i = 0; i < 16; ++i) {
    int c = lane * 16 + i;
    out[(size_t)b * D_ + c] = (v[i] - mean) * inv * g[c] + bt[c];
  }
}

// ---------------- host ----------------
extern "C" void kernel_launch(void* const* d_in, const int* in_sizes, int n_in,
                              void* d_out, int out_size, void* d_ws, size_t ws_size,
                              hipStream_t stream) {
  const int*   text  = (const int*)d_in[0];
  const float* emb   = (const float*)d_in[1];
  const float* cls   = (const float*)d_in[2];
  const float* clsg  = (const float*)d_in[3];
  const float* clsb  = (const float*)d_in[4];
  const float* lng   = (const float*)d_in[5];
  const float* lnb   = (const float*)d_in[6];
  const float* wfused = (const float*)d_in[7];
  const float* wattn  = (const float*)d_in[8];
  const float* wff    = (const float*)d_in[9];
  (void)in_sizes; (void)n_in; (void)out_size; (void)ws_size;

  uint8_t* w = (uint8_t*)d_ws;
  size_t off = 0;
  auto alloc = [&](size_t bytes) -> void* {
    void* p = w + off;
    off += (bytes + 255) & ~(size_t)255;
    return p;
  };
  // total workspace ~201 MB
  float* x       = (float*)alloc((size_t)MTOK * D_ * 4);
  bf16* h_bf     = (bf16*)alloc((size_t)MP * D_ * 2);
  bf16* proj_bf  = (bf16*)alloc((size_t)MTOK * FUSED_ * 2);
  bf16* q_bf     = (bf16*)alloc((size_t)B_ * H_ * QP * DH_ * 2);
  bf16* k_bf     = (bf16*)alloc((size_t)B_ * KP * DH_ * 2);
  bf16* vT_bf    = (bf16*)alloc((size_t)B_ * DH_ * KP * 2);
  bf16* ffh_bf   = (bf16*)alloc((size_t)MP * FF_ * 2);
  bf16* o_bf     = (bf16*)alloc((size_t)MP * (H_ * DH_) * 2);
  bf16* wfused_b = (bf16*)alloc((size_t)L_ * D_ * FUSED_ * 2);
  bf16* wattn_b  = (bf16*)alloc((size_t)L_ * D_ * D_ * 2);
  bf16* wff_b    = (bf16*)alloc((size_t)L_ * FF_ * D_ * 2);
  float* cosT    = (float*)alloc((size_t)N_ * 32 * 4);
  float* sinT    = (float*)alloc((size_t)N_ * 32 * 4);

  auto cdiv = [](long a, long b) { return (int)((a + b - 1) / b); };

  k_f32_to_bf16<<<cdiv((long)L_ * D_ * FUSED_, 256), 256, 0, stream>>>(wfused, wfused_b, L_ * D_ * FUSED_);
  k_f32_to_bf16<<<cdiv((long)L_ * D_ * D_, 256), 256, 0, stream>>>(wattn, wattn_b, L_ * D_ * D_);
  k_f32_to_bf16<<<cdiv((long)L_ * FF_ * D_, 256), 256, 0, stream>>>(wff, wff_b, L_ * FF_ * D_);
  k_rope_tables<<<cdiv((long)N_ * 32, 256), 256, 0, stream>>>(cosT, sinT);
  k_embed<<<cdiv((long)MTOK * D_, 256), 256, 0, stream>>>(text, emb, cls, x);

  for (int i = 0; i < L_; ++i) {
    k_ln_bf16<<<cdiv(MTOK, 8), 256, 0, stream>>>(x, lng + i * D_, lnb + i * D_, h_bf);
    dim3 g1(FUSED_ / 128, MP / 128);
    k_gemm<1><<<g1, 256, 0, stream>>>(h_bf, wfused_b + (size_t)i * D_ * FUSED_, proj_bf,
                                      MTOK, FUSED_, D_);
    k_post_proj<<<MTOK, 256, 0, stream>>>(proj_bf, q_bf, k_bf, vT_bf, ffh_bf, cosT, sinT);
    k_attn<<<cdiv(B_ * H_ * QT, 8), 256, 0, stream>>>(q_bf, k_bf, vT_bf, text, o_bf);
    dim3 g2(D_ / 128, MP / 128);
    k_gemm<2><<<g2, 256, 0, stream>>>(o_bf, wattn_b + (size_t)i * D_ * D_, x, MTOK, D_, D_);
    k_gemm<2><<<g2, 256, 0, stream>>>(ffh_bf, wff_b + (size_t)i * FF_ * D_, x, MTOK, D_, FF_);
  }
  k_copy_tokens<<<cdiv((long)B_ * S_ * D_, 256), 256, 0, stream>>>(x, (float*)d_out);
  k_cls_ln<<<1, 128, 0, stream>>>(x, clsg, clsb, (float*)d_out + (size_t)B_ * S_ * D_);
}
